// AttentionLayer_41592463294777
// MI455X (gfx1250) — matrix operand order repaired and checked by reading, not compiled, not run on hardware
//
#include <hip/hip_runtime.h>
#include <math.h>

typedef __attribute__((ext_vector_type(16))) __bf16 v16bf;
typedef __attribute__((ext_vector_type(8)))  float  v8f;

#define B_    64
#define T_    4096
#define D_    256
#define CTX_  100
#define CTXP  112      // CTX padded to 7 x 16 WMMA N-tiles
#define NT    7        // CTXP / 16
#define ROWS_PER_BLOCK 256
#define THREADS_A      512
#define CH    16       // T-chunks for weighted-sum partials
#define TCH   (T_ / CH)

// -------- Kernel 1: logits[b,t] = sum_c u[c] * tanh( (X W)[b,t,c] ) --------
// bf16 WMMA 16x16x32, f32 accumulate; W staged in LDS transposed+padded.
__global__ __launch_bounds__(THREADS_A) void logits_kernel(
    const float* __restrict__ X, const float* __restrict__ W,
    const float* __restrict__ u, float* __restrict__ logits)
{
  __shared__ __bf16 Wt[CTXP][D_];   // [n][k], 57344 bytes
  __shared__ float  uS[CTXP];

  const int tid = threadIdx.x;

  // Stage W^T (bf16, zero-padded to 112 cols) and u into LDS.
  for (int i = tid; i < CTXP * D_; i += THREADS_A) {
    const int n = i >> 8;          // 0..111
    const int k = i & (D_ - 1);    // 0..255
    const float w = (n < CTX_) ? W[k * CTX_ + n] : 0.0f;
    Wt[n][k] = (__bf16)w;
  }
  if (tid < CTXP) uS[tid] = (tid < CTX_) ? u[tid] : 0.0f;
  __syncthreads();

  const int lane = tid & 31;
  const int wave = tid >> 5;
  const int b    = blockIdx.y;
  const int m0   = blockIdx.x * ROWS_PER_BLOCK + wave * 16;   // T-strip base

  // A-fragment addressing, contiguous k order (same as B below):
  // lanes 0-15 : M=lane,   K = ko+0..15
  // lanes 16-31: M=lane-16,K = ko+16..31
  const int row  = m0 + (lane & 15);
  const int ksel = (lane < 16) ? 0 : 16;
  const float* __restrict__ xp = X + ((size_t)b * T_ + row) * D_;

  const v8f vzero = {};
  v8f acc[NT];
#pragma unroll
  for (int nt = 0; nt < NT; ++nt) acc[nt] = vzero;

  for (int ko = 0; ko < D_; ko += 32) {
    const float* g = xp + ko + ksel;
    const float4 f0 = *(const float4*)(g);        // k .. k+3
    const float4 f1 = *(const float4*)(g + 4);    // k+4 .. k+7
    const float4 f2 = *(const float4*)(g + 8);    // k+8 .. k+11
    const float4 f3 = *(const float4*)(g + 12);   // k+12 .. k+15

    v16bf a;
    a[0]  = (__bf16)f0.x; a[1]  = (__bf16)f0.y; a[2]  = (__bf16)f0.z; a[3]  = (__bf16)f0.w;
    a[4]  = (__bf16)f1.x; a[5]  = (__bf16)f1.y; a[6]  = (__bf16)f1.z; a[7]  = (__bf16)f1.w;
    a[8]  = (__bf16)f2.x; a[9]  = (__bf16)f2.y; a[10] = (__bf16)f2.z; a[11] = (__bf16)f2.w;
    a[12] = (__bf16)f3.x; a[13] = (__bf16)f3.y; a[14] = (__bf16)f3.z; a[15] = (__bf16)f3.w;

    // B-fragment, same contiguous k order as A:
    // lanes 0-15 : N=lane,    K = ko+0..15  (contiguous 32B in Wt row)
    // lanes 16-31: N=lane-16, K = ko+16..31
    const int kB = ko + ((lane < 16) ? 0 : 16);
#pragma unroll
    for (int nt = 0; nt < NT; ++nt) {
      const v16bf bm = *(const v16bf*)(&Wt[nt * 16 + (lane & 15)][kB]);
      acc[nt] = __builtin_amdgcn_wmma_f32_16x16x32_bf16(
          false, a, false, bm, (short)0, acc[nt], false, false);
    }
  }

  // Epilogue: D-fragment lane L(<16) vgpr r -> (M=r, N=L); lane L+16 -> (M=8+r).
  // Each lane owns one CTX column per tile; apply tanh * u[c], reduce over 16 lanes.
  float part[8];
#pragma unroll
  for (int r = 0; r < 8; ++r) part[r] = 0.0f;
#pragma unroll
  for (int nt = 0; nt < NT; ++nt) {
    const float uc = uS[nt * 16 + (lane & 15)];
#pragma unroll
    for (int r = 0; r < 8; ++r)
      part[r] += tanhf(acc[nt][r]) * uc;
  }
#pragma unroll
  for (int r = 0; r < 8; ++r) {
    part[r] += __shfl_xor(part[r], 1, 32);
    part[r] += __shfl_xor(part[r], 2, 32);
    part[r] += __shfl_xor(part[r], 4, 32);
    part[r] += __shfl_xor(part[r], 8, 32);
  }
  if ((lane & 15) == 0) {
    const int rbase = m0 + (lane >> 4) * 8;   // lane 0 -> rows 0..7, lane 16 -> rows 8..15
#pragma unroll
    for (int r = 0; r < 8; ++r)
      logits[(size_t)b * T_ + rbase + r] = part[r];
  }
}

// -------- Kernel 2: in-place softmax over T for each batch --------
__global__ __launch_bounds__(256) void softmax_kernel(float* __restrict__ att)
{
  __shared__ float red[256];
  const int b   = blockIdx.x;
  const int tid = threadIdx.x;
  float* row = att + (size_t)b * T_;

  float m = -3.402823466e+38f;
#pragma unroll
  for (int i = 0; i < T_ / 256; ++i) m = fmaxf(m, row[tid + i * 256]);
  red[tid] = m;
  __syncthreads();
  for (int s = 128; s > 0; s >>= 1) {
    if (tid < s) red[tid] = fmaxf(red[tid], red[tid + s]);
    __syncthreads();
  }
  m = red[0];
  __syncthreads();

  float vals[T_ / 256];
  float sum = 0.0f;
#pragma unroll
  for (int i = 0; i < T_ / 256; ++i) {
    const float e = __expf(row[tid + i * 256] - m);
    vals[i] = e;
    sum += e;
  }
  red[tid] = sum;
  __syncthreads();
  for (int s = 128; s > 0; s >>= 1) {
    if (tid < s) red[tid] += red[tid + s];
    __syncthreads();
  }
  const float inv = 1.0f / red[0];
#pragma unroll
  for (int i = 0; i < T_ / 256; ++i) row[tid + i * 256] = vals[i] * inv;
}

// -------- Kernel 3: partial weighted sums over T-chunks (deterministic) --------
__global__ __launch_bounds__(256) void wsum_partial_kernel(
    const float* __restrict__ X, const float* __restrict__ att,
    float* __restrict__ partial)
{
  __shared__ float aS[TCH];
  const int b   = blockIdx.y;
  const int ch  = blockIdx.x;
  const int tid = threadIdx.x;   // = d column
  const int t0  = ch * TCH;

  aS[tid] = att[(size_t)b * T_ + t0 + tid];
  __syncthreads();

  const float* __restrict__ xp = X + ((size_t)b * T_ + t0) * D_ + tid;
  float acc = 0.0f;
  for (int t = 0; t < TCH; ++t) {
    __builtin_prefetch(xp + (t + 8) * D_, 0, 0);   // global_prefetch_b8 on stream
    acc = fmaf(xp[t * D_], aS[t], acc);
  }
  partial[((size_t)b * CH + ch) * D_ + tid] = acc;
}

// -------- Kernel 4: reduce partials -> out (B, D) --------
__global__ __launch_bounds__(256) void wsum_reduce_kernel(
    const float* __restrict__ partial, float* __restrict__ out)
{
  const int b = blockIdx.x;
  const int d = threadIdx.x;
  float acc = 0.0f;
#pragma unroll
  for (int ch = 0; ch < CH; ++ch)
    acc += partial[((size_t)b * CH + ch) * D_ + d];
  out[(size_t)b * D_ + d] = acc;
}

extern "C" void kernel_launch(void* const* d_in, const int* in_sizes, int n_in,
                              void* d_out, int out_size, void* d_ws, size_t ws_size,
                              hipStream_t stream)
{
  (void)in_sizes; (void)n_in; (void)out_size; (void)ws_size;
  const float* X = (const float*)d_in[0];
  const float* W = (const float*)d_in[1];
  const float* u = (const float*)d_in[2];
  float* out = (float*)d_out;

  float* logits  = (float*)d_ws;                      // B*T floats (reused in-place as att)
  float* partial = logits + (size_t)B_ * T_;          // B*CH*D floats

  dim3 gA(T_ / ROWS_PER_BLOCK, B_);                   // (16, 64)
  logits_kernel<<<gA, THREADS_A, 0, stream>>>(X, W, u, logits);
  softmax_kernel<<<B_, 256, 0, stream>>>(logits);
  dim3 gC(CH, B_);                                    // (16, 64)
  wsum_partial_kernel<<<gC, 256, 0, stream>>>(X, logits, partial);
  wsum_reduce_kernel<<<B_, 256, 0, stream>>>(partial, out);
}
